// GNN_GNN_Model_61649960567381
// MI455X (gfx1250) — matrix operand order repaired and checked by reading, not compiled, not run on hardware
//
#include <hip/hip_runtime.h>
#include <stdint.h>

// ---------------------------------------------------------------------------
// Types for CDNA5 WMMA (wave32)
// ---------------------------------------------------------------------------
typedef __attribute__((ext_vector_type(16))) __bf16 v16bf;
typedef __attribute__((ext_vector_type(8)))  float  v8f;

union AFrag { uint4 u[2]; v16bf v; };

__device__ __forceinline__ unsigned short f2bf(float f) {
  unsigned int u = __float_as_uint(f);
  unsigned int r = u + 0x7FFFu + ((u >> 16) & 1u);   // round-to-nearest-even
  return (unsigned short)(r >> 16);
}

// monotone int encoding of float for atomicMax-based segment max
__device__ __forceinline__ int enc_f(float f) {
  int i = __float_as_int(f);
  return (i >= 0) ? i : (i ^ 0x7FFFFFFF);
}
__device__ __forceinline__ float dec_f(int i) {
  return __int_as_float((i >= 0) ? i : (i ^ 0x7FFFFFFF));
}

// ---------------------------------------------------------------------------
// Pack f32 row-major [M x K] into bf16 WMMA A-fragment tiles.
// Tile (mt,kt) = 512 halves at Ap[(mt*Kt+kt)*512].
// Lane layout (ISA 7.12.2, 16-bit A 16x32):
//   element j of lane L holds A[mt*16 + (L&15)][kt*32 + kloc],
//   kloc = (j<8 ? j : j+8) + 8*(L>=16)
// Storage: half[(j>>3)*256 + L*8 + (j&7)]  -> GEMM does 2x contiguous b128/lane.
// ---------------------------------------------------------------------------
__global__ void pack_a_kernel(const float* __restrict__ A,
                              unsigned short* __restrict__ Ap, int M, int K) {
  int Kt = K >> 5;
  long long tid = (long long)blockIdx.x * blockDim.x + threadIdx.x;
  long long total = (long long)(M >> 4) * Kt * 512;
  if (tid >= total) return;
  int local = (int)(tid & 511);
  long long tile = tid >> 9;
  int mt = (int)(tile / Kt), kt = (int)(tile % Kt);
  int ki = local & 31;   // coalesced over K
  int mi = local >> 5;
  float v = A[(long long)(mt * 16 + mi) * K + (kt * 32 + ki)];
  int lane = mi | (((ki >> 3) & 1) << 4);
  int j    = (((ki >> 4) & 1) << 3) | (ki & 7);
  Ap[tile * 512 + ((j >> 3) << 8) + (lane << 3) + (j & 7)] = f2bf(v);
}

// ---------------------------------------------------------------------------
// Pack f32 row-major [K x N] into bf16 WMMA B-fragment tiles.
// Tile (nt,kt) = 512 halves at Bp[(nt*Kt+kt)*512].
// B 32x16 16-bit: same k order as A: k = (j<8 ? j : j+8) + 8*(L>=16).
// ---------------------------------------------------------------------------
__global__ void pack_b_kernel(const float* __restrict__ B,
                              unsigned short* __restrict__ Bp, int K, int N) {
  int Kt = K >> 5;
  long long tid = (long long)blockIdx.x * blockDim.x + threadIdx.x;
  long long total = (long long)(N >> 4) * Kt * 512;
  if (tid >= total) return;
  int local = (int)(tid & 511);
  long long tile = tid >> 9;
  int nt = (int)(tile / Kt), kt = (int)(tile % Kt);
  int ni = local & 15;
  int ki = local >> 4;   // 0..31
  float v = B[(long long)(kt * 32 + ki) * N + (nt * 16 + ni)];
  int lane = ni | (((ki >> 3) & 1) << 4);
  int j    = (((ki >> 4) & 1) << 3) | (ki & 7);
  Bp[tile * 512 + ((j >> 3) << 8) + (lane << 3) + (j & 7)] = f2bf(v);
}

// ---------------------------------------------------------------------------
// Register-blocked WMMA GEMM: C[M x N] = Apacked @ Bpacked (+ bias), f32 out.
// One wave computes a 32(M) x 64(N) super-tile = 2x4 grid of 16x16 WMMA tiles:
// per K-step, 6 fragment loads feed 8 v_wmma -> ~21 flop/byte from L2.
// Requires Mt even and Nt % 4 == 0 (true for all shapes here).
// ---------------------------------------------------------------------------
__global__ void wmma_gemm_kernel(const unsigned short* __restrict__ Ap,
                                 const unsigned short* __restrict__ Bp,
                                 const float* __restrict__ bias,
                                 float* __restrict__ C,
                                 int Mb, int Nb, int Kt, int N) {
  int wave = threadIdx.x >> 5;
  int lane = threadIdx.x & 31;
  int t = blockIdx.x * (blockDim.x >> 5) + wave;
  if (t >= Mb * Nb) return;          // wave-uniform exit; EXEC stays all-ones
  int mb = t / Nb, nb = t % Nb;

  const uint4* pa0 = (const uint4*)(Ap) + (long long)(mb * 2 + 0) * Kt * 64;
  const uint4* pa1 = (const uint4*)(Ap) + (long long)(mb * 2 + 1) * Kt * 64;
  const uint4* pb0 = (const uint4*)(Bp) + (long long)(nb * 4 + 0) * Kt * 64;
  const uint4* pb1 = (const uint4*)(Bp) + (long long)(nb * 4 + 1) * Kt * 64;
  const uint4* pb2 = (const uint4*)(Bp) + (long long)(nb * 4 + 2) * Kt * 64;
  const uint4* pb3 = (const uint4*)(Bp) + (long long)(nb * 4 + 3) * Kt * 64;

  v8f acc[2][4];
#pragma unroll
  for (int mi = 0; mi < 2; ++mi)
#pragma unroll
    for (int ni = 0; ni < 4; ++ni)
      acc[mi][ni] = (v8f){0.f, 0.f, 0.f, 0.f, 0.f, 0.f, 0.f, 0.f};

  for (int kt = 0; kt < Kt; ++kt) {
    AFrag a0, a1, b0, b1, b2, b3;
    a0.u[0] = pa0[lane]; a0.u[1] = pa0[32 + lane];
    a1.u[0] = pa1[lane]; a1.u[1] = pa1[32 + lane];
    b0.u[0] = pb0[lane]; b0.u[1] = pb0[32 + lane];
    b1.u[0] = pb1[lane]; b1.u[1] = pb1[32 + lane];
    b2.u[0] = pb2[lane]; b2.u[1] = pb2[32 + lane];
    b3.u[0] = pb3[lane]; b3.u[1] = pb3[32 + lane];
    pa0 += 64; pa1 += 64; pb0 += 64; pb1 += 64; pb2 += 64; pb3 += 64;

    acc[0][0] = __builtin_amdgcn_wmma_f32_16x16x32_bf16(false, a0.v, false, b0.v, (short)0, acc[0][0], false, false);
    acc[0][1] = __builtin_amdgcn_wmma_f32_16x16x32_bf16(false, a0.v, false, b1.v, (short)0, acc[0][1], false, false);
    acc[0][2] = __builtin_amdgcn_wmma_f32_16x16x32_bf16(false, a0.v, false, b2.v, (short)0, acc[0][2], false, false);
    acc[0][3] = __builtin_amdgcn_wmma_f32_16x16x32_bf16(false, a0.v, false, b3.v, (short)0, acc[0][3], false, false);
    acc[1][0] = __builtin_amdgcn_wmma_f32_16x16x32_bf16(false, a1.v, false, b0.v, (short)0, acc[1][0], false, false);
    acc[1][1] = __builtin_amdgcn_wmma_f32_16x16x32_bf16(false, a1.v, false, b1.v, (short)0, acc[1][1], false, false);
    acc[1][2] = __builtin_amdgcn_wmma_f32_16x16x32_bf16(false, a1.v, false, b2.v, (short)0, acc[1][2], false, false);
    acc[1][3] = __builtin_amdgcn_wmma_f32_16x16x32_bf16(false, a1.v, false, b3.v, (short)0, acc[1][3], false, false);
  }

#pragma unroll
  for (int mi = 0; mi < 2; ++mi) {
    int row0 = (mb * 2 + mi) * 16 + ((lane >> 4) << 3);
#pragma unroll
    for (int ni = 0; ni < 4; ++ni) {
      int col = (nb * 4 + ni) * 16 + (lane & 15);
      float bv = bias ? bias[col] : 0.0f;
#pragma unroll
      for (int r = 0; r < 8; ++r)
        C[(long long)(row0 + r) * N + col] = acc[mi][ni][r] + bv;
    }
  }
}

// ---------------------------------------------------------------------------
// Elementwise helpers
// ---------------------------------------------------------------------------
__global__ void fill_f32_kernel(float* p, float v, int n) {
  int i = blockIdx.x * blockDim.x + threadIdx.x;
  if (i < n) p[i] = v;
}
__global__ void fill_i32_kernel(int* p, int v, int n) {
  int i = blockIdx.x * blockDim.x + threadIdx.x;
  if (i < n) p[i] = v;
}
__global__ void add_bias2_kernel(float* __restrict__ X, const float* __restrict__ b1,
                                 const float* __restrict__ b2, long long total, int C) {
  long long i = (long long)blockIdx.x * blockDim.x + threadIdx.x;
  if (i >= total) return;
  int c = (int)(i % C);
  float v = b1[c];
  if (b2) v += b2[c];
  X[i] += v;
}
__global__ void relu_kernel(const float* __restrict__ in, float* __restrict__ out,
                            long long n) {
  long long i = (long long)blockIdx.x * blockDim.x + threadIdx.x;
  if (i < n) out[i] = fmaxf(in[i], 0.0f);
}

// ---------------------------------------------------------------------------
// GATv2 edge kernels (edges e < E from arrays, e >= E are self loops)
// ---------------------------------------------------------------------------
__global__ void gat_logit_kernel(const float* __restrict__ xl, const float* __restrict__ xr,
                                 const int* __restrict__ src, const int* __restrict__ dst,
                                 int E, int Nself, int C, const float* __restrict__ att,
                                 float* __restrict__ logit, int* __restrict__ mmax) {
  int w = (int)(((long long)blockIdx.x * blockDim.x + threadIdx.x) >> 5);
  int lane = threadIdx.x & 31;
  if (w >= E + Nself) return;
  int s, d;
  if (w < E) { s = src[w]; d = dst[w]; } else { s = d = w - E; }
  float acc = 0.0f;
  for (int c = lane; c < C; c += 32) {
    float v = xl[(long long)s * C + c] + xr[(long long)d * C + c];
    v = (v > 0.0f) ? v : 0.2f * v;                 // leaky_relu(0.2)
    acc += v * att[c];
  }
  for (int off = 16; off; off >>= 1) acc += __shfl_xor(acc, off, 32);
  if (lane == 0) {
    logit[w] = acc;
    atomicMax(&mmax[d], enc_f(acc));
  }
}

__global__ void gat_expsum_kernel(const float* __restrict__ logit,
                                  const int* __restrict__ dst, int E, int Nself,
                                  const int* __restrict__ mmax,
                                  float* __restrict__ ex, float* __restrict__ ssum) {
  int e = blockIdx.x * blockDim.x + threadIdx.x;
  if (e >= E + Nself) return;
  int d = (e < E) ? dst[e] : (e - E);
  float v = __expf(logit[e] - dec_f(mmax[d]));
  ex[e] = v;
  atomicAdd(&ssum[d], v);
}

__global__ void gat_alpha_kernel(float* __restrict__ ex, const int* __restrict__ dst,
                                 int E, int Nself, const float* __restrict__ ssum) {
  int e = blockIdx.x * blockDim.x + threadIdx.x;
  if (e >= E + Nself) return;
  int d = (e < E) ? dst[e] : (e - E);
  ex[e] = ex[e] / ssum[d];
}

__global__ void gat_scatter_kernel(const float* __restrict__ xl,
                                   const int* __restrict__ src, const int* __restrict__ dst,
                                   int E, int Nself, int C,
                                   const float* __restrict__ alpha, float* __restrict__ out) {
  long long tid = (long long)blockIdx.x * blockDim.x + threadIdx.x;
  long long total = (long long)(E + Nself) * C;
  if (tid >= total) return;
  int e = (int)(tid / C);
  int c = (int)(tid % C);
  int s, d;
  if (e < E) { s = src[e]; d = dst[e]; } else { s = d = e - E; }
  atomicAdd(&out[(long long)d * C + c], alpha[e] * xl[(long long)s * C + c]);
}

// ---------------------------------------------------------------------------
// GCN kernels
// ---------------------------------------------------------------------------
__global__ void gcn_deg_kernel(const int* __restrict__ dst, int E, int Nself,
                               float* __restrict__ deg) {
  int e = blockIdx.x * blockDim.x + threadIdx.x;
  if (e >= E + Nself) return;
  int d = (e < E) ? dst[e] : (e - E);
  atomicAdd(&deg[d], 1.0f);
}
__global__ void gcn_dinv_kernel(const float* __restrict__ deg, float* __restrict__ dinv,
                                int n) {
  int i = blockIdx.x * blockDim.x + threadIdx.x;
  if (i >= n) return;
  float g = deg[i];
  dinv[i] = (g > 0.0f) ? rsqrtf(fmaxf(g, 1e-12f)) : 0.0f;
}
__global__ void gcn_scatter_kernel(const float* __restrict__ xw,
                                   const int* __restrict__ src, const int* __restrict__ dst,
                                   int E, int Nself, int C,
                                   const float* __restrict__ dinv, float* __restrict__ out) {
  long long tid = (long long)blockIdx.x * blockDim.x + threadIdx.x;
  long long total = (long long)(E + Nself) * C;
  if (tid >= total) return;
  int e = (int)(tid / C);
  int c = (int)(tid % C);
  int s, d;
  if (e < E) { s = src[e]; d = dst[e]; } else { s = d = e - E; }
  atomicAdd(&out[(long long)d * C + c],
            dinv[s] * dinv[d] * xw[(long long)s * C + c]);
}

// ---------------------------------------------------------------------------
// Link predictor: out[i] = dot(g2[ig[i]], c2[ic[i]]) over C=128
// ---------------------------------------------------------------------------
__global__ void link_dot_kernel(const float* __restrict__ g2, const float* __restrict__ c2,
                                const int* __restrict__ ig, const int* __restrict__ ic,
                                int NL, int C, float* __restrict__ out) {
  int w = (int)(((long long)blockIdx.x * blockDim.x + threadIdx.x) >> 5);
  int lane = threadIdx.x & 31;
  if (w >= NL) return;
  long long gi = (long long)ig[w] * C, ci = (long long)ic[w] * C;
  float acc = 0.0f;
  for (int c = lane; c < C; c += 32) acc += g2[gi + c] * c2[ci + c];
  for (int off = 16; off; off >>= 1) acc += __shfl_xor(acc, off, 32);
  if (lane == 0) out[w] = acc;
}

// ---------------------------------------------------------------------------
// Host orchestration
// ---------------------------------------------------------------------------
extern "C" void kernel_launch(void* const* d_in, const int* in_sizes, int n_in,
                              void* d_out, int out_size, void* d_ws, size_t ws_size,
                              hipStream_t stream) {
  (void)n_in; (void)ws_size;
  const int FG = 1024, FC = 2048, EMB = 256, H1 = 256, H2 = 128;
  const int NG = in_sizes[0] / FG;
  const int NC = in_sizes[1] / FC;
  const int E_GG = in_sizes[42] / 2;
  const int E_CC = in_sizes[43] / 2;
  const int NL = out_size;

  // ---- inputs (flattened setup_inputs dict order) ----
  const float* x_gene   = (const float*)d_in[0];
  const float* x_cell   = (const float*)d_in[1];
  const float* Wg_in    = (const float*)d_in[2];
  const float* bg_in    = (const float*)d_in[3];
  const float* Wc_in    = (const float*)d_in[4];
  const float* bc_in    = (const float*)d_in[5];
  const float* g1f_Wl   = (const float*)d_in[6];
  const float* g1f_bl   = (const float*)d_in[7];
  const float* g1f_Wr   = (const float*)d_in[8];
  const float* g1f_br   = (const float*)d_in[9];
  const float* g1f_att  = (const float*)d_in[10];
  const float* g1f_b    = (const float*)d_in[11];
  const float* g1r_Wl   = (const float*)d_in[12];
  const float* g1r_bl   = (const float*)d_in[13];
  const float* g1r_Wr   = (const float*)d_in[14];
  const float* g1r_br   = (const float*)d_in[15];
  const float* g1r_att  = (const float*)d_in[16];
  const float* g1r_b    = (const float*)d_in[17];
  const float* gcn1_W   = (const float*)d_in[18];
  const float* gcn1_b   = (const float*)d_in[19];
  const float* g2f_Wl   = (const float*)d_in[20];
  const float* g2f_bl   = (const float*)d_in[21];
  const float* g2f_Wr   = (const float*)d_in[22];
  const float* g2f_br   = (const float*)d_in[23];
  const float* g2f_att  = (const float*)d_in[24];
  const float* g2f_b    = (const float*)d_in[25];
  const float* g2r_Wl   = (const float*)d_in[26];
  const float* g2r_bl   = (const float*)d_in[27];
  const float* g2r_Wr   = (const float*)d_in[28];
  const float* g2r_br   = (const float*)d_in[29];
  const float* g2r_att  = (const float*)d_in[30];
  const float* g2r_b    = (const float*)d_in[31];
  const float* gcn2_W   = (const float*)d_in[32];
  const float* gcn2_b   = (const float*)d_in[33];
  const float* sk_g1_W  = (const float*)d_in[34];
  const float* sk_g1_b  = (const float*)d_in[35];
  const float* sk_g2_W  = (const float*)d_in[36];
  const float* sk_g2_b  = (const float*)d_in[37];
  const float* sk_c1_W  = (const float*)d_in[38];
  const float* sk_c1_b  = (const float*)d_in[39];
  const float* sk_c2_W  = (const float*)d_in[40];
  const float* sk_c2_b  = (const float*)d_in[41];
  const int* eg0 = (const int*)d_in[42];            // edge_gg src row
  const int* eg1 = eg0 + E_GG;                      // edge_gg dst row
  const int* ec0 = (const int*)d_in[43];
  const int* ec1 = ec0 + E_CC;
  const int* ell_gene = (const int*)d_in[44];
  const int* ell_cell = (const int*)d_in[45];
  float* out = (float*)d_out;

  // ---- workspace bump allocator ----
  size_t off = 0;
  auto alloc = [&](size_t bytes) -> char* {
    off = (off + 255) & ~(size_t)255;
    char* p = (char*)d_ws + off;
    off += bytes;
    return p;
  };
  auto allocF = [&](size_t n) { return (float*)alloc(n * 4); };
  auto allocH = [&](size_t n) { return (unsigned short*)alloc(n * 2); };
  auto allocI = [&](size_t n) { return (int*)alloc(n * 4); };

  // packed bf16 weights (K*N halves each)
  unsigned short* WgP    = allocH((size_t)FG * EMB);
  unsigned short* WcP    = allocH((size_t)FC * EMB);
  unsigned short* g1fWlP = allocH((size_t)EMB * H1);
  unsigned short* g1fWrP = allocH((size_t)EMB * H1);
  unsigned short* g1rWlP = allocH((size_t)EMB * H1);
  unsigned short* g1rWrP = allocH((size_t)EMB * H1);
  unsigned short* gcn1P  = allocH((size_t)EMB * H1);
  unsigned short* skg1P  = allocH((size_t)EMB * H1);
  unsigned short* skc1P  = allocH((size_t)EMB * H1);
  unsigned short* g2fWlP = allocH((size_t)H1 * H2);
  unsigned short* g2fWrP = allocH((size_t)H1 * H2);
  unsigned short* g2rWlP = allocH((size_t)H1 * H2);
  unsigned short* g2rWrP = allocH((size_t)H1 * H2);
  unsigned short* gcn2P  = allocH((size_t)H1 * H2);
  unsigned short* skg2P  = allocH((size_t)H1 * H2);
  unsigned short* skc2P  = allocH((size_t)H1 * H2);

  unsigned short* xgP = allocH((size_t)NG * FG);
  unsigned short* xcP = allocH((size_t)NC * FC);
  float* g0  = allocF((size_t)NG * EMB);
  float* c0  = allocF((size_t)NC * EMB);
  unsigned short* g0P = allocH((size_t)NG * EMB);
  unsigned short* c0P = allocH((size_t)NC * EMB);
  float* xl  = allocF((size_t)NG * H1);       // reused for layer2 (H2 <= H1)
  float* xr  = allocF((size_t)NG * H1);
  float* xwc = allocF((size_t)NC * H1);
  float* g1  = allocF((size_t)NG * H1);
  float* c1  = allocF((size_t)NC * H1);
  float* ga  = allocF((size_t)NG * H1);
  float* ca  = allocF((size_t)NC * H1);
  unsigned short* g1P = allocH((size_t)NG * H1);
  unsigned short* gaP = allocH((size_t)NG * H1);
  unsigned short* c1P = allocH((size_t)NC * H1);
  unsigned short* caP = allocH((size_t)NC * H1);
  float* g2  = allocF((size_t)NG * H2);
  float* c2  = allocF((size_t)NC * H2);
  float* logit = allocF((size_t)E_GG + NG);
  float* exb   = allocF((size_t)E_GG + NG);
  int*   mmax  = allocI((size_t)NG);
  float* ssum  = allocF((size_t)NG);
  float* deg   = allocF((size_t)NC);
  float* dinv  = allocF((size_t)NC);

  // ---- launch helpers ----
  auto packA = [&](const float* A, unsigned short* Ap, int M, int K) {
    long long total = (long long)(M >> 4) * (K >> 5) * 512;
    pack_a_kernel<<<(unsigned)((total + 255) / 256), 256, 0, stream>>>(A, Ap, M, K);
  };
  auto packB = [&](const float* B, unsigned short* Bp, int K, int N) {
    long long total = (long long)(N >> 4) * (K >> 5) * 512;
    pack_b_kernel<<<(unsigned)((total + 255) / 256), 256, 0, stream>>>(B, Bp, K, N);
  };
  auto gemm = [&](const unsigned short* Ap, const unsigned short* Bp,
                  const float* bias, float* C, int M, int N, int K) {
    int Mb = M >> 5, Nb = N >> 6, Kt = K >> 5;   // 32x64 super-tiles per wave
    wmma_gemm_kernel<<<(unsigned)((Mb * Nb + 7) / 8), 256, 0, stream>>>(
        Ap, Bp, bias, C, Mb, Nb, Kt, N);
  };
  auto fillF = [&](float* p, float v, int n) {
    fill_f32_kernel<<<(unsigned)((n + 255) / 256), 256, 0, stream>>>(p, v, n);
  };
  auto fillI = [&](int* p, int v, int n) {
    fill_i32_kernel<<<(unsigned)((n + 255) / 256), 256, 0, stream>>>(p, v, n);
  };
  auto addBias2 = [&](float* X, const float* b1, const float* b2, int M, int C) {
    long long total = (long long)M * C;
    add_bias2_kernel<<<(unsigned)((total + 255) / 256), 256, 0, stream>>>(X, b1, b2, total, C);
  };
  auto relu = [&](const float* in, float* o, long long n) {
    relu_kernel<<<(unsigned)((n + 255) / 256), 256, 0, stream>>>(in, o, n);
  };

  // GATv2 relation: accumulate segment-softmax aggregation into `accum`
  auto runGat = [&](const unsigned short* xP, int din, int dout, int Nn,
                    const int* src, const int* dst, int E,
                    const unsigned short* WlP, const float* bl,
                    const unsigned short* WrP, const float* br,
                    const float* att, float* accum) {
    gemm(xP, WlP, bl, xl, Nn, dout, din);
    gemm(xP, WrP, br, xr, Nn, dout, din);
    fillI(mmax, (int)0x80000000, Nn);
    int Et = E + Nn;
    gat_logit_kernel<<<(unsigned)(((long long)Et * 32 + 255) / 256), 256, 0, stream>>>(
        xl, xr, src, dst, E, Nn, dout, att, logit, mmax);
    fillF(ssum, 0.0f, Nn);
    gat_expsum_kernel<<<(unsigned)((Et + 255) / 256), 256, 0, stream>>>(
        logit, dst, E, Nn, mmax, exb, ssum);
    gat_alpha_kernel<<<(unsigned)((Et + 255) / 256), 256, 0, stream>>>(
        exb, dst, E, Nn, ssum);
    long long total = (long long)Et * dout;
    gat_scatter_kernel<<<(unsigned)((total + 255) / 256), 256, 0, stream>>>(
        xl, src, dst, E, Nn, dout, exb, accum);
  };

  // ---- 0) pack all weights + inputs ----
  packB(Wg_in, WgP, FG, EMB);
  packB(Wc_in, WcP, FC, EMB);
  packB(g1f_Wl, g1fWlP, EMB, H1);  packB(g1f_Wr, g1fWrP, EMB, H1);
  packB(g1r_Wl, g1rWlP, EMB, H1);  packB(g1r_Wr, g1rWrP, EMB, H1);
  packB(gcn1_W, gcn1P, EMB, H1);
  packB(sk_g1_W, skg1P, EMB, H1);  packB(sk_c1_W, skc1P, EMB, H1);
  packB(g2f_Wl, g2fWlP, H1, H2);   packB(g2f_Wr, g2fWrP, H1, H2);
  packB(g2r_Wl, g2rWlP, H1, H2);   packB(g2r_Wr, g2rWrP, H1, H2);
  packB(gcn2_W, gcn2P, H1, H2);
  packB(sk_g2_W, skg2P, H1, H2);   packB(sk_c2_W, skc2P, H1, H2);
  packA(x_gene, xgP, NG, FG);
  packA(x_cell, xcP, NC, FC);

  // ---- 1) input projections ----
  gemm(xgP, WgP, bg_in, g0, NG, EMB, FG);
  gemm(xcP, WcP, bc_in, c0, NC, EMB, FC);
  packA(g0, g0P, NG, EMB);
  packA(c0, c0P, NC, EMB);

  // ---- 2) layer 1 genes: skip + two GATv2 relations ----
  gemm(g0P, skg1P, sk_g1_b, g1, NG, H1, EMB);
  addBias2(g1, g1f_b, g1r_b, NG, H1);
  runGat(g0P, EMB, H1, NG, eg0, eg1, E_GG, g1fWlP, g1f_bl, g1fWrP, g1f_br, g1f_att, g1);
  runGat(g0P, EMB, H1, NG, eg1, eg0, E_GG, g1rWlP, g1r_bl, g1rWrP, g1r_br, g1r_att, g1);

  // ---- 3) layer 1 cells: skip + GCN ----
  gemm(c0P, skc1P, sk_c1_b, c1, NC, H1, EMB);
  addBias2(c1, gcn1_b, nullptr, NC, H1);
  gemm(c0P, gcn1P, nullptr, xwc, NC, H1, EMB);
  fillF(deg, 0.0f, NC);
  int EtC = E_CC + NC;
  gcn_deg_kernel<<<(unsigned)((EtC + 255) / 256), 256, 0, stream>>>(ec1, E_CC, NC, deg);
  gcn_dinv_kernel<<<(unsigned)((NC + 255) / 256), 256, 0, stream>>>(deg, dinv, NC);
  {
    long long total = (long long)EtC * H1;
    gcn_scatter_kernel<<<(unsigned)((total + 255) / 256), 256, 0, stream>>>(
        xwc, ec0, ec1, E_CC, NC, H1, dinv, c1);
  }

  // ---- 4) activations + repack ----
  relu(g1, ga, (long long)NG * H1);
  relu(c1, ca, (long long)NC * H1);
  packA(g1, g1P, NG, H1);
  packA(ga, gaP, NG, H1);
  packA(c1, c1P, NC, H1);
  packA(ca, caP, NC, H1);

  // ---- 5) layer 2 genes ----
  gemm(g1P, skg2P, sk_g2_b, g2, NG, H2, H1);
  addBias2(g2, g2f_b, g2r_b, NG, H2);
  runGat(gaP, H1, H2, NG, eg0, eg1, E_GG, g2fWlP, g2f_bl, g2fWrP, g2f_br, g2f_att, g2);
  runGat(gaP, H1, H2, NG, eg1, eg0, E_GG, g2rWlP, g2r_bl, g2rWrP, g2r_br, g2r_att, g2);

  // ---- 6) layer 2 cells (dinv reused: same edge set) ----
  gemm(c1P, skc2P, sk_c2_b, c2, NC, H2, H1);
  addBias2(c2, gcn2_b, nullptr, NC, H2);
  gemm(caP, gcn2P, nullptr, xwc, NC, H2, H1);
  {
    long long total = (long long)EtC * H2;
    gcn_scatter_kernel<<<(unsigned)((total + 255) / 256), 256, 0, stream>>>(
        xwc, ec0, ec1, E_CC, NC, H2, dinv, c2);
  }

  // ---- 7) link predictor ----
  link_dot_kernel<<<(unsigned)(((long long)NL * 32 + 255) / 256), 256, 0, stream>>>(
      g2, c2, ell_gene, ell_cell, NL, H2, out);
}
